// DKT_PEBG_33775622815756
// MI455X (gfx1250) — matrix operand order repaired and checked by reading, not compiled, not run on hardware
//
#include <hip/hip_runtime.h>
#include <hip/hip_bf16.h>
#include <math.h>

// ---------------------------------------------------------------------------
// DKT-PEBG forward for MI455X (gfx1250, wave32, WMMA + TDM)
//   K0: one-shot f32 -> bf16 conversion of pro_embed / W_ih / W_hh
//   K1: xg = (masked embed) @ W_ih^T + b_ih           (bf16 WMMA, f32 accum)
//   K2: 200-step LSTM scan, single WGP, W_hh in LDS via TENSOR_LOAD_TO_LDS
//   K3: per-(b,t) gathered output dot + sigmoid       (wave reduction)
// ---------------------------------------------------------------------------

typedef __attribute__((ext_vector_type(16))) __bf16   bf16x16;
typedef __attribute__((ext_vector_type(8)))  __bf16   bf16x8;
typedef __attribute__((ext_vector_type(8)))  float    f32x8;
typedef __attribute__((ext_vector_type(4)))  float    f32x4;
typedef __attribute__((ext_vector_type(4)))  unsigned u32x4;
typedef __attribute__((ext_vector_type(8)))  int      i32x8;
typedef __attribute__((ext_vector_type(4)))  int      i32x4;

#define B_SZ   64
#define S_SZ   200
#define EMBED  128
#define HID    128
#define NGATE  512          // 4*HID
#define KIN    256          // 2*EMBED
#define NTOK   (B_SZ * S_SZ)
#define WROW   136          // padded LDS row (bf16): 272B stride -> conflict-free

static __device__ __forceinline__ bf16x16 join8(bf16x8 lo, bf16x8 hi) {
  return __builtin_shufflevector(lo, hi, 0,1,2,3,4,5,6,7,8,9,10,11,12,13,14,15);
}
// CDNA5 has V_TANH_F32 in the transcendental pipe (co-executes with XDL WMMA).
static __device__ __forceinline__ float tanh_fast(float x) {
#if __has_builtin(__builtin_amdgcn_tanhf)
  return __builtin_amdgcn_tanhf(x);
#else
  return tanhf(x);
#endif
}
static __device__ __forceinline__ float rcp_fast(float x) {
#if __has_builtin(__builtin_amdgcn_rcpf)
  return __builtin_amdgcn_rcpf(x);
#else
  return 1.0f / x;
#endif
}
static __device__ __forceinline__ float sigmoidf_(float x) {
  return rcp_fast(1.0f + __expf(-x));
}

// ---------------------------------------------------------------------------
// K0: f32 -> bf16, 4 elems/thread (n is always a multiple of 4).
// ---------------------------------------------------------------------------
__global__ __launch_bounds__(256)
void k0_cvt_bf16(const float* __restrict__ src, __bf16* __restrict__ dst, int n)
{
  const int i4 = (blockIdx.x * 256 + threadIdx.x) * 4;
  if (i4 >= n) return;
  const f32x4 v = *(const f32x4*)(src + i4);
  __bf16 o[4];
  #pragma unroll
  for (int j = 0; j < 4; ++j) o[j] = (__bf16)v[j];
  *(unsigned long long*)(dst + i4) = *(unsigned long long*)o;   // 8B store
}

// ---------------------------------------------------------------------------
// K1: one 16x16 output tile per wave. M=12800, N=512, K=256.
// Pure bf16 16B loads; y-mask applied with vector selects (no converts).
// ---------------------------------------------------------------------------
__global__ __launch_bounds__(256)
void k1_embed_gemm(const int* __restrict__ X, const int* __restrict__ Y,
                   const __bf16* __restrict__ PEb, const __bf16* __restrict__ WIb,
                   const float* __restrict__ b_ih, float* __restrict__ XG)
{
  const int wave = threadIdx.x >> 5;
  const int lane = threadIdx.x & 31;
  const int tile = blockIdx.x * 8 + wave;   // 800*32 = 25600 tiles
  const int m_tile = tile >> 5;             // 0..799
  const int n_tile = tile & 31;             // 0..31
  const int half = lane >> 4;
  const int l16  = lane & 15;

  const int token = m_tile * 16 + l16;
  const int ei    = X[token];
  const int yv    = Y[token];               // -1, 0, 1
  const __bf16* erow = PEb + (size_t)ei * EMBED;

  const int n = n_tile * 16 + l16;
  const __bf16* wrow = WIb + (size_t)n * KIN;

  f32x8 acc = {};
  const bf16x8 zero8 = {};
  #pragma unroll
  for (int k0 = 0; k0 < KIN; k0 += 32) {
    // A fragment: lanes<16 hold K=k0+[0..7],[16..23]; lanes>=16 +8
    const int  sel = k0 >> 7;               // which 128-half of K
    const bool use = (yv == sel);
    const int  ka  = (k0 & 127) + half * 8;
    bf16x8 e0 = *(const bf16x8*)(erow + ka);
    bf16x8 e1 = *(const bf16x8*)(erow + ka + 16);
    e0 = use ? e0 : zero8;
    e1 = use ? e1 : zero8;
    const bf16x16 a = join8(e0, e1);

    // B fragment: lanes<16 hold K=k0+[0..7],[16..23]; lanes>=16 +8
    const int kb = k0 + half * 8;
    const bf16x16 b = join8(*(const bf16x8*)(wrow + kb),
                            *(const bf16x8*)(wrow + kb + 16));

    acc = __builtin_amdgcn_wmma_f32_16x16x32_bf16(false, a, false, b,
                                                  (short)0, acc, false, false);
  }

  const float bias = b_ih[n];
  #pragma unroll
  for (int r = 0; r < 8; ++r) {
    const int row = m_tile * 16 + r + half * 8;       // token index (b*S + s)
    XG[(size_t)row * NGATE + n] = acc[r] + bias;
  }
}

// ---------------------------------------------------------------------------
// K2: persistent single-workgroup LSTM scan. 1024 threads = 32 waves.
// W_hh (bf16) staged into padded LDS by ONE tensor_load_to_lds DMA:
//   D# group0: count=1, lds_addr=0, global_addr=Whh_bf, type=2
//   D# group1: data_size=2B, tile 128x512, tensor_dim0(_stride)=128,
//              pad_enable, pad_interval=64 DW (one row), pad_amount=4 DW
// XG fragments for step t+1 are loaded right after step t's gate math.
// ---------------------------------------------------------------------------
__global__ __launch_bounds__(1024)
void k2_lstm(const float* __restrict__ XG, const __bf16* __restrict__ Whh_bf,
             const float* __restrict__ b_hh, float* __restrict__ HS,
             u32x4 tdm_g0, i32x8 tdm_g1)
{
  extern __shared__ __bf16 smem[];
  __bf16* whh = smem;                       // [512][WROW]
  __bf16* hbf = smem + NGATE * WROW;        // [64][WROW]

  const int tid  = threadIdx.x;
  const int wave = tid >> 5;
  const int lane = tid & 31;
  const int half = lane >> 4;
  const int l16  = lane & 15;

#if __has_builtin(__builtin_amdgcn_tensor_load_to_lds)
  if (wave == 0) {
    const i32x4 z4 = {};
    const i32x8 z8 = {};
    __builtin_amdgcn_tensor_load_to_lds(tdm_g0, tdm_g1, z4, z4, z8, 0);
    __builtin_amdgcn_s_wait_tensorcnt(0);
  }
  (void)Whh_bf;
#else
  (void)tdm_g0; (void)tdm_g1;
  for (int idx = tid; idx < NGATE * HID; idx += 1024)
    whh[(idx >> 7) * WROW + (idx & 127)] = Whh_bf[idx];
#endif
  for (int idx = tid; idx < B_SZ * WROW; idx += 1024) hbf[idx] = (__bf16)0.0f;
  __syncthreads();

  const int m_tile  = wave >> 3;            // 0..3
  const int col_blk = wave & 7;             // 0..7
  const int row0 = m_tile * 16;
  const int col0 = col_blk * 16;
  const int arow = row0 + l16;              // A-matrix row (batch index)

  float bias_g[4];
  #pragma unroll
  for (int g = 0; g < 4; ++g) bias_g[g] = b_hh[g * HID + col0 + l16];

  float cstate[8];
  #pragma unroll
  for (int r = 0; r < 8; ++r) cstate[r] = 0.0f;

  // Preload XG fragments for t = 0.
  f32x8 acc[4];
  #pragma unroll
  for (int g = 0; g < 4; ++g) {
    const int n = g * HID + col0 + l16;
    #pragma unroll
    for (int r = 0; r < 8; ++r)
      acc[g][r] = XG[((size_t)(row0 + r + half * 8) * S_SZ) * NGATE + n];
  }

  for (int t = 0; t < S_SZ; ++t) {
    // GEMM: acc += h_{t-1} @ W_hh^T   (K = 128, 16 wmma)
    #pragma unroll
    for (int k0 = 0; k0 < HID; k0 += 32) {
      const int ka = k0 + half * 8;
      const bf16x16 a = join8(*(const bf16x8*)&hbf[arow * WROW + ka],
                              *(const bf16x8*)&hbf[arow * WROW + ka + 16]);
      #pragma unroll
      for (int g = 0; g < 4; ++g) {
        const int n  = g * HID + col0 + l16;
        const int kb = k0 + half * 8;
        const bf16x16 bb = join8(*(const bf16x8*)&whh[n * WROW + kb],
                                 *(const bf16x8*)&whh[n * WROW + kb + 16]);
        acc[g] = __builtin_amdgcn_wmma_f32_16x16x32_bf16(false, a, false, bb,
                                                         (short)0, acc[g], false, false);
      }
    }

    // Gates: i, f, g, o  ->  c, h   (acc dead afterwards)
    float hnew[8];
    #pragma unroll
    for (int r = 0; r < 8; ++r) {
      const float gi = sigmoidf_(acc[0][r] + bias_g[0]);
      const float gf = sigmoidf_(acc[1][r] + bias_g[1]);
      const float gg = tanh_fast(acc[2][r] + bias_g[2]);
      const float go = sigmoidf_(acc[3][r] + bias_g[3]);
      const float c  = gf * cstate[r] + gi * gg;
      cstate[r] = c;
      hnew[r]   = go * tanh_fast(c);
    }

    // Issue next step's XG loads now; latency hides behind stores + barriers.
    const int tn = (t + 1 < S_SZ) ? (t + 1) : t;
    #pragma unroll
    for (int g = 0; g < 4; ++g) {
      const int n = g * HID + col0 + l16;
      #pragma unroll
      for (int r = 0; r < 8; ++r)
        acc[g][r] = XG[((size_t)(row0 + r + half * 8) * S_SZ + tn) * NGATE + n];
    }

    __syncthreads();                        // everyone done reading h_{t-1}
    #pragma unroll
    for (int r = 0; r < 8; ++r) {
      const int b = row0 + r + half * 8;
      const int c = col0 + l16;
      hbf[b * WROW + c] = (__bf16)hnew[r];                    // next step's A
      HS[((size_t)b * S_SZ + t) * HID + c] = hnew[r];         // f32 history
    }
    __syncthreads();                        // h_t visible to all waves
  }
}

// ---------------------------------------------------------------------------
// K3: one wave per (b, t) pair: dot(h[b,t,:], W_out[max(X[b,t+1]-1,0),:]).
// ---------------------------------------------------------------------------
__global__ __launch_bounds__(256)
void k3_outgather(const int* __restrict__ X, const float* __restrict__ HS,
                  const float* __restrict__ W_out, const float* __restrict__ b_out,
                  float* __restrict__ out)
{
  const int wave = threadIdx.x >> 5;
  const int lane = threadIdx.x & 31;
  const int pair = blockIdx.x * 8 + wave;
  if (pair >= B_SZ * (S_SZ - 1)) return;
  const int b = pair / (S_SZ - 1);
  const int t = pair % (S_SZ - 1);

  const int idx = X[b * S_SZ + t + 1];
  const int row = (idx > 1) ? (idx - 1) : 0;        // clip(idx-1, 0)
  const float* h = HS    + ((size_t)b * S_SZ + t) * HID;
  const float* w = W_out + (size_t)row * HID;

  float sum = 0.0f;
  #pragma unroll
  for (int j = 0; j < 4; ++j) sum += h[lane + 32 * j] * w[lane + 32 * j];
  #pragma unroll
  for (int m = 16; m >= 1; m >>= 1) sum += __shfl_xor(sum, m, 32);

  if (lane == 0)
    out[pair] = (idx == 0) ? 0.0f : sigmoidf_(sum + b_out[row]);
}

// ---------------------------------------------------------------------------
extern "C" void kernel_launch(void* const* d_in, const int* in_sizes, int n_in,
                              void* d_out, int out_size, void* d_ws, size_t ws_size,
                              hipStream_t stream)
{
  (void)in_sizes; (void)n_in; (void)out_size; (void)ws_size;
  const int*   X    = (const int*)  d_in[0];
  const int*   Y    = (const int*)  d_in[1];
  const float* PE   = (const float*)d_in[2];
  const float* Wih  = (const float*)d_in[3];
  const float* Whh  = (const float*)d_in[4];
  const float* bih  = (const float*)d_in[5];
  const float* bhh  = (const float*)d_in[6];
  const float* Wout = (const float*)d_in[7];
  const float* bout = (const float*)d_in[8];
  float* out = (float*)d_out;

  // Workspace layout (16B aligned partitions)
  char* ws = (char*)d_ws;
  float*  XG     = (float*)ws;                                   ws += (size_t)NTOK * NGATE * 4;   // 26.2 MB
  float*  HS     = (float*)ws;                                   ws += (size_t)NTOK * HID * 4;     //  6.6 MB
  __bf16* Whh_bf = (__bf16*)ws;                                  ws += (size_t)NGATE * HID * 2;    //  128 KB
  __bf16* Wih_bf = (__bf16*)ws;                                  ws += (size_t)NGATE * KIN * 2;    //  256 KB
  __bf16* PE_bf  = (__bf16*)ws;                                  // 10000*128*2 = 2.5 MB

  // K0: one-shot weight/embedding conversion to bf16
  k0_cvt_bf16<<<(NGATE * HID  / 4 + 255) / 256, 256, 0, stream>>>(Whh, Whh_bf, NGATE * HID);
  k0_cvt_bf16<<<(NGATE * KIN  / 4 + 255) / 256, 256, 0, stream>>>(Wih, Wih_bf, NGATE * KIN);
  k0_cvt_bf16<<<(10000 * EMBED / 4 + 255) / 256, 256, 0, stream>>>(PE, PE_bf, 10000 * EMBED);

  // K1: 25600 tiles, 8 waves/block
  k1_embed_gemm<<<(800 * 32) / 8, 256, 0, stream>>>(X, Y, PE_bf, Wih_bf, bih, XG);

  // TDM descriptor for W_hh bf16 -> padded LDS (built on host; uniform SGPR args)
  const unsigned long long wa = (unsigned long long)(uintptr_t)Whh_bf;
  u32x4 g0 = {};
  g0[0] = 1u;                                            // count=1, no gather
  g0[1] = 0u;                                            // lds_addr = 0 (whh at LDS base)
  g0[2] = (unsigned)(wa & 0xFFFFFFFFu);                  // global_addr[31:0]
  g0[3] = (unsigned)((wa >> 32) & 0x01FFFFFFu) | (2u << 30);  // addr[56:32] | type=2
  i32x8 g1 = {};
  g1[0] = (1 << 16)                                      // data_size = 2B
        | (1 << 20)                                      // pad_enable
        | (5 << 22)                                      // pad_interval: 64 DW = one 128-elem row
        | (3 << 25);                                     // pad_amount: 4 DW = 8 bf16
  g1[1] = (int)(128u << 16);                             // tensor_dim0 = 128 (bits 79:48)
  g1[2] = (int)(512u << 16);                             // tensor_dim1 = 512 (bits 111:80)
  g1[3] = (int)(128u << 16);                             // tile_dim0 = 128 (bits 127:112)
  g1[4] = 512;                                           // tile_dim1 = 512 (bits 143:128)
  g1[5] = 128;                                           // tensor_dim0_stride = 128 (bits 207:160)
  g1[6] = 0; g1[7] = 0;

  // K2: single WGP-resident workgroup; 156,672 B dynamic LDS (>64KB -> raise cap)
  const size_t lds = (size_t)(NGATE + B_SZ) * WROW * sizeof(__bf16);
  (void)hipFuncSetAttribute((const void*)k2_lstm,
                            hipFuncAttributeMaxDynamicSharedMemorySize, (int)lds);
  k2_lstm<<<1, 1024, lds, stream>>>(XG, Whh_bf, bhh, HS, g0, g1);

  // K3: 64*199 = 12736 (b,t) pairs, 8 waves/block
  k3_outgather<<<(B_SZ * (S_SZ - 1) + 7) / 8, 256, 0, stream>>>(X, HS, Wout, bout, out);
}
